// AttentiveFPREG_68083821576343
// MI455X (gfx1250) — matrix operand order repaired and checked by reading, not compiled, not run on hardware
//
#include <hip/hip_runtime.h>
#include <hip/hip_bf16.h>
#include <math.h>
#include <stdint.h>

#define H 128
#define WPB 8  // waves per block (wave32)

typedef __attribute__((ext_vector_type(16))) __bf16 bf16x16;
typedef __attribute__((ext_vector_type(8)))  float  f32x8;

union FragB { bf16x16 v; uint4 q[2]; };

__device__ __forceinline__ uint16_t cvt_bf16(float f) {
    uint32_t u = __float_as_uint(f);
    uint32_t r = (u + 0x7FFFu + ((u >> 16) & 1u)) >> 16;
    return (uint16_t)r;
}
__device__ __forceinline__ float bf16f(uint16_t h) {
    return __uint_as_float(((uint32_t)h) << 16);
}
__device__ __forceinline__ float leaky01(float v) { return v > 0.f ? v : 0.01f * v; }
__device__ __forceinline__ void atomicMaxF(float* a, float v) {
    if (v >= 0.f) atomicMax((int*)a, __float_as_int(v));
    else          atomicMin((unsigned int*)a, __float_as_uint(v));
}

// ---------------------------------------------------------------------------
// Generic bf16 WMMA GEMM, K fixed = 128.
//   A: bf16 [M x 128], rows optionally gathered through gidx
//   Bt: bf16 [Nout x 128]  (pre-transposed B so columns are contiguous)
//   epilogue: +bias[n], +r1coef[m]*r1vec[n], leaky, then either
//     - store Out fp32 / OutBf bf16 (optionally non-temporal), or
//     - atomicAdd into ScatOut[sidx[m]*Nout + n] scaled by scale[m]
// One 16x16 tile per wave; block = 8 waves covering 128 output columns.
// The 16x128 A tile is moved global->LDS with GLOBAL_LOAD_ASYNC_TO_LDS_B128
// (ASYNCcnt-tracked TDM-class async copy; no VGPR round trip).
// ---------------------------------------------------------------------------
__global__ __launch_bounds__(256)
void k_wmma_gemm(const uint16_t* __restrict__ A, const int* __restrict__ gidx, int M,
                 const uint16_t* __restrict__ Bt, int Nout,
                 const float* __restrict__ bias,
                 const float* __restrict__ r1coef, const float* __restrict__ r1vec,
                 int act, int ntOut,
                 float* __restrict__ Out, uint16_t* __restrict__ OutBf,
                 float* __restrict__ ScatOut, const int* __restrict__ sidx,
                 const float* __restrict__ scale)
{
    __shared__ __align__(16) uint16_t As[16 * H];
    const int t = threadIdx.x;
    const int wave = t >> 5, lane = t & 31, half = lane >> 4, l15 = lane & 15;
    const int m0 = blockIdx.x * 16;
    const int n0 = (blockIdx.y * WPB + wave) * 16;   // Nout is a multiple of 128

    {   // async A-tile copy: 16 rows x 128 bf16 = one b128 per thread, straight to LDS
        const int lrow = t >> 4, lchunk = t & 15;
        const int gm = m0 + lrow;
        int grow = 0;
        if (gm < M) grow = gidx ? gidx[gm] : gm;
        const uint16_t* gp = A + (size_t)grow * H + lchunk * 8;
        uint32_t ldsoff = (uint32_t)(uintptr_t)(As + lrow * H + lchunk * 8);
        uint64_t ga = (uint64_t)(uintptr_t)gp;
        asm volatile("global_load_async_to_lds_b128 %0, %1, off"
                     :: "v"(ldsoff), "v"(ga) : "memory");
        asm volatile("s_wait_asynccnt 0x0" ::: "memory");
    }
    __syncthreads();

    // A frag: lane l15 = row; half 0 -> K {0..7,16..23}, half 1 -> K {8..15,24..31}
    const uint16_t* Ab = As + l15 * H + half * 8;
    // B frag: lane l15 = col n0+l15; half 0 -> K {0..7,16..23}, half 1 -> K {8..15,24..31}
    const uint16_t* Bb = Bt + (size_t)(n0 + l15) * H + half * 8;
    __builtin_prefetch(Bb, 0, 1);

    f32x8 c = {0.f, 0.f, 0.f, 0.f, 0.f, 0.f, 0.f, 0.f};
#pragma unroll
    for (int kk = 0; kk < H; kk += 32) {
        FragB a, b;
        a.q[0] = *(const uint4*)(Ab + kk);
        a.q[1] = *(const uint4*)(Ab + kk + 16);
        b.q[0] = *(const uint4*)(Bb + kk);
        b.q[1] = *(const uint4*)(Bb + kk + 16);
        c = __builtin_amdgcn_wmma_f32_16x16x32_bf16(false, a.v, false, b.v,
                                                    (short)0, c, false, false);
    }

    const int n = n0 + l15;   // C/D layout: lane = column, vgpr j = row (+8 upper half)
    if (ScatOut) {
#pragma unroll
        for (int j = 0; j < 8; ++j) {
            const int m = m0 + (half << 3) + j;
            if (m < M)
                atomicAdd(&ScatOut[(size_t)sidx[m] * Nout + n], c[j] * scale[m]);
        }
    } else {
        const float bias_n = bias  ? bias[n]  : 0.f;   // hoisted: uniform over j
        const float r1_n   = r1vec ? r1vec[n] : 0.f;
#pragma unroll
        for (int j = 0; j < 8; ++j) {
            const int m = m0 + (half << 3) + j;
            if (m >= M) continue;
            float v = c[j] + bias_n;
            if (r1vec) v += r1coef[m] * r1_n;
            if (act)   v = leaky01(v);
            if (Out) {
                float* p = &Out[(size_t)m * Nout + n];
                if (ntOut) __builtin_nontemporal_store(v, p); else *p = v;
            }
            if (OutBf) {
                uint16_t bv = cvt_bf16(v);
                uint16_t* p = &OutBf[(size_t)m * Nout + n];
                if (ntOut) __builtin_nontemporal_store(bv, p); else *p = bv;
            }
        }
    }
}

// ---------------------------------------------------------------------------
// helper kernels
// ---------------------------------------------------------------------------
__global__ void k_fill(float* p, float v, int n) {
    int i = blockIdx.x * 256 + threadIdx.x; if (i < n) p[i] = v;
}
__global__ void k_cvt(const float* __restrict__ x, uint16_t* __restrict__ y, int n) {
    int i = blockIdx.x * 256 + threadIdx.x; if (i < n) y[i] = cvt_bf16(x[i]);
}
__global__ void k_trans128(const float* __restrict__ W, uint16_t* __restrict__ Bt) {
    // W: [128x128] row-major -> Bt[n*128+k] = W[k*128+n]
    int i = blockIdx.x * 256 + threadIdx.x; if (i >= 128 * 128) return;
    int k = i >> 7, n = i & 127;
    Bt[n * 128 + k] = cvt_bf16(W[i]);
}
// one wave per row: d1[m] = X[m]·v1, d2[m] = X[m]·v2 (v2 optional)
__global__ void k_rowdot2(const float* __restrict__ X, int M,
                          const float* __restrict__ v1, const float* __restrict__ v2,
                          float* __restrict__ d1, float* __restrict__ d2) {
    int row = blockIdx.x * WPB + (threadIdx.x >> 5);
    int lane = threadIdx.x & 31;
    if (row >= M) return;
    float4 x = *(const float4*)(X + (size_t)row * H + lane * 4);
    float4 a = *(const float4*)(v1 + lane * 4);
    float s1 = x.x * a.x + x.y * a.y + x.z * a.z + x.w * a.w;
    float s2 = 0.f;
    if (v2) { float4 b = *(const float4*)(v2 + lane * 4);
              s2 = x.x * b.x + x.y * b.y + x.z * b.z + x.w * b.w; }
#pragma unroll
    for (int o = 16; o; o >>= 1) { s1 += __shfl_xor(s1, o, 32); s2 += __shfl_xor(s2, o, 32); }
    if (lane == 0) { d1[row] = s1; if (d2) d2[row] = s2; }
}
// alpha[e] = leaky(hj[e]·attL + xr[dst[e]])    (hj read non-temporally: streamed)
__global__ void k_gate_alpha(const uint16_t* __restrict__ hj, int E,
                             const float* __restrict__ attL,
                             const float* __restrict__ xr, const int* __restrict__ dst,
                             float* __restrict__ alpha) {
    int e = blockIdx.x * WPB + (threadIdx.x >> 5);
    int lane = threadIdx.x & 31;
    if (e >= E) return;
    const uint16_t* hp = hj + (size_t)e * H + lane * 4;
    float s = 0.f;
#pragma unroll
    for (int i = 0; i < 4; ++i)
        s += bf16f(__builtin_nontemporal_load(hp + i)) * attL[lane * 4 + i];
#pragma unroll
    for (int o = 16; o; o >>= 1) s += __shfl_xor(s, o, 32);
    if (lane == 0) alpha[e] = leaky01(s + xr[dst[e]]);
}
__global__ void k_edge_alpha(const float* dA, const float* dB,
                             const int* src, const int* dst, int E, float* alpha) {
    int e = blockIdx.x * 256 + threadIdx.x; if (e >= E) return;
    alpha[e] = leaky01(dA[src[e]] + dB[dst[e]]);
}
__global__ void k_node_alpha(const float* asrc, const float* ad,
                             const int* batch, int N, float* alpha) {
    int i = blockIdx.x * 256 + threadIdx.x; if (i >= N) return;
    alpha[i] = leaky01(asrc[i] + ad[batch[i]]);
}
__global__ void k_seg_max(const float* a, const int* idx, int M, float* smax) {
    int i = blockIdx.x * 256 + threadIdx.x; if (i >= M) return;
    atomicMaxF(&smax[idx[i]], a[i]);
}
__global__ void k_seg_expsum(float* a, const int* idx, int M,
                             const float* smax, float* ssum) {
    int i = blockIdx.x * 256 + threadIdx.x; if (i >= M) return;
    float m = smax[idx[i]];
    if (m < -3.0e38f) m = 0.f;            // empty segment -> max was -inf
    float e = expf(a[i] - m);
    a[i] = e;
    atomicAdd(&ssum[idx[i]], e);
}
__global__ void k_seg_norm(float* a, const int* idx, int M, const float* ssum) {
    int i = blockIdx.x * 256 + threadIdx.x; if (i >= M) return;
    a[i] = a[i] / (ssum[idx[i]] + 1e-16f);
}
// h[dst[e]] += alpha[e] * hs[src[e]]   (one wave per edge, 4 cols per lane)
__global__ void k_edge_scatter(const float* __restrict__ hs, const float* __restrict__ alpha,
                               const int* __restrict__ src, const int* __restrict__ dst,
                               int E, float* __restrict__ hout) {
    int tid = blockIdx.x * 256 + threadIdx.x;
    int e = tid >> 5, lane = tid & 31;
    if (e >= E) return;
    float a = alpha[e];
    float4 v = *(const float4*)(hs + (size_t)src[e] * H + lane * 4);
    float* o = hout + (size_t)dst[e] * H + lane * 4;
    atomicAdd(o + 0, a * v.x); atomicAdd(o + 1, a * v.y);
    atomicAdd(o + 2, a * v.z); atomicAdd(o + 3, a * v.w);
}
// out[seg[r]] += (alpha ? alpha[r] : 1) * X[r]
__global__ void k_seg_scatter(const float* __restrict__ X, const float* __restrict__ alpha,
                              const int* __restrict__ seg, int M, float* __restrict__ out) {
    int tid = blockIdx.x * 256 + threadIdx.x;
    int r = tid >> 5, lane = tid & 31;
    if (r >= M) return;
    float a = alpha ? alpha[r] : 1.f;
    float4 v = *(const float4*)(X + (size_t)r * H + lane * 4);
    float* o = out + (size_t)seg[r] * H + lane * 4;
    atomicAdd(o + 0, a * v.x); atomicAdd(o + 1, a * v.y);
    atomicAdd(o + 2, a * v.z); atomicAdd(o + 3, a * v.w);
}
__global__ void k_elu_bias(float* __restrict__ h, const float* __restrict__ bias,
                           int Mrows, uint16_t* __restrict__ hbf) {
    int i = blockIdx.x * 256 + threadIdx.x; if (i >= Mrows * H) return;
    int col = i & (H - 1);
    float v = h[i] + bias[col];
    v = v > 0.f ? v : expm1f(v);
    h[i] = v;
    if (hbf) hbf[i] = cvt_bf16(v);
}
__global__ void k_relu_bf(float* x, int n, uint16_t* xbf) {
    int i = blockIdx.x * 256 + threadIdx.x; if (i >= n) return;
    float v = fmaxf(x[i], 0.f);
    x[i] = v; if (xbf) xbf[i] = cvt_bf16(v);
}
// fused GRU gate combine + relu; X is the hidden state (updated in place)
__global__ void k_gru(const float* __restrict__ gi, const float* __restrict__ gh,
                      float* __restrict__ X, uint16_t* __restrict__ Xbf, int M) {
    int i = blockIdx.x * 256 + threadIdx.x; if (i >= M * H) return;
    int row = i >> 7, col = i & 127;
    size_t b = (size_t)row * 384;
    float ir = gi[b + col], iz = gi[b + 128 + col], inn = gi[b + 256 + col];
    float hr = gh[b + col], hz = gh[b + 128 + col], hn = gh[b + 256 + col];
    float r = 1.f / (1.f + expf(-(ir + hr)));
    float z = 1.f / (1.f + expf(-(iz + hz)));
    float nn = tanhf(inn + r * hn);
    float v = (1.f - z) * nn + z * X[i];
    v = fmaxf(v, 0.f);
    X[i] = v; if (Xbf) Xbf[i] = cvt_bf16(v);
}
__global__ void k_matvec128(const float* __restrict__ W, const float* __restrict__ v,
                            float* __restrict__ wv) {
    int k = threadIdx.x; if (k >= 128) return;
    float s = 0.f;
    for (int n = 0; n < 128; ++n) s += W[k * 128 + n] * v[n];
    wv[k] = s;
}
__global__ void k_head(const float* __restrict__ out, int G,
                       const float* __restrict__ W2, const float* __restrict__ b2,
                       const float* __restrict__ Wp1, const float* __restrict__ bp1,
                       const float* __restrict__ Wp2, const float* __restrict__ bp2,
                       float* __restrict__ res) {
    int g = blockIdx.x * 256 + threadIdx.x; if (g >= G) return;
    const float* o = out + (size_t)g * H;
    float o0 = b2[0], o1 = b2[1];
    for (int k = 0; k < H; ++k) { float v = o[k]; o0 += v * W2[k * 2]; o1 += v * W2[k * 2 + 1]; }
    float p = fmaxf(o0 * Wp1[0] + o1 * Wp1[1] + bp1[0], 0.f);
    res[g] = p * Wp2[0] + bp2[0];
}

// ---------------------------------------------------------------------------
extern "C" void kernel_launch(void* const* d_in, const int* in_sizes, int n_in,
                              void* d_out, int out_size, void* d_ws, size_t ws_size,
                              hipStream_t stream) {
    const float* x    = (const float*)d_in[0];
    const int*   ei   = (const int*)d_in[1];
    const int*   batch= (const int*)d_in[2];
    const float* eatt = (const float*)d_in[3];
    const float* W1   = (const float*)d_in[4];
    const float* b1   = (const float*)d_in[5];
    const float* Wg1  = (const float*)d_in[6];
    const float* attL = (const float*)d_in[7];
    const float* attR = (const float*)d_in[8];
    const float* Wg2  = (const float*)d_in[9];
    const float* bg   = (const float*)d_in[10];
    const float* gWi0 = (const float*)d_in[11];
    const float* gWh0 = (const float*)d_in[12];
    const float* gbi0 = (const float*)d_in[13];
    const float* gbh0 = (const float*)d_in[14];
    const float* Wc   = (const float*)d_in[15];
    const float* aS   = (const float*)d_in[16];
    const float* aD   = (const float*)d_in[17];
    const float* bc   = (const float*)d_in[18];
    const float* gWi  = (const float*)d_in[19];
    const float* gWh  = (const float*)d_in[20];
    const float* gbi  = (const float*)d_in[21];
    const float* gbh  = (const float*)d_in[22];
    const float* Wm   = (const float*)d_in[23];
    const float* aSm  = (const float*)d_in[24];
    const float* aDm  = (const float*)d_in[25];
    const float* bm   = (const float*)d_in[26];
    const float* mWi  = (const float*)d_in[27];
    const float* mWh  = (const float*)d_in[28];
    const float* mbi  = (const float*)d_in[29];
    const float* mbh  = (const float*)d_in[30];
    const float* W2   = (const float*)d_in[31];
    const float* b2   = (const float*)d_in[32];
    const float* Wp1  = (const float*)d_in[33];
    const float* bp1  = (const float*)d_in[34];
    const float* Wp2  = (const float*)d_in[35];
    const float* bp2  = (const float*)d_in[36];
    (void)n_in; (void)ws_size;

    const int N = in_sizes[0] / H;
    const int E = in_sizes[3];
    const int G = out_size;
    const int NL1 = 3, NT = 2;
    const int* src = ei;
    const int* dst = ei + E;
    const float* wrow = Wg1 + 128 * 128;   // edge_attr row of Wg1, fp32

    // ---- workspace bump allocator ----
    char* ws = (char*)d_ws;
    size_t off = 0;
    auto alloc = [&](size_t bytes) -> void* {
        void* p = ws + off;
        off = (off + bytes + 255) & ~(size_t)255;
        return p;
    };
    float*    xf     = (float*)   alloc((size_t)N * H * 4);
    uint16_t* xbf    = (uint16_t*)alloc((size_t)N * H * 2);
    float*    hnode  = (float*)   alloc((size_t)N * H * 4);
    uint16_t* hbf    = (uint16_t*)alloc((size_t)N * H * 2);
    float*    hsbuf  = (float*)   alloc((size_t)N * H * 4);
    float*    dA     = (float*)   alloc((size_t)N * 4);
    float*    dB     = (float*)   alloc((size_t)N * 4);
    float*    asrc   = (float*)   alloc((size_t)N * 4);
    float*    segmax = (float*)   alloc((size_t)N * 4);
    float*    segsum = (float*)   alloc((size_t)N * 4);
    float*    alphaE = (float*)   alloc((size_t)E * 4);
    float*    outG   = (float*)   alloc((size_t)G * H * 4);
    uint16_t* outGbf = (uint16_t*)alloc((size_t)G * H * 2);
    float*    hG     = (float*)   alloc((size_t)G * H * 4);
    uint16_t* hGbf   = (uint16_t*)alloc((size_t)G * H * 2);
    float*    adG    = (float*)   alloc((size_t)G * 4);
    float*    wv     = (float*)   alloc((size_t)H * 4);
    float*    giG    = (float*)   alloc((size_t)G * 384 * 4);
    float*    ghG    = (float*)   alloc((size_t)G * 384 * 4);
    uint16_t* W1t    = (uint16_t*)alloc(H * H * 2);
    uint16_t* Wg1at  = (uint16_t*)alloc(H * H * 2);
    uint16_t* Wg2t   = (uint16_t*)alloc(H * H * 2);
    uint16_t* gWi0b  = (uint16_t*)alloc(384 * H * 2);
    uint16_t* gWh0b  = (uint16_t*)alloc(384 * H * 2);
    uint16_t* Wct    = (uint16_t*)alloc((size_t)NL1 * H * H * 2);
    uint16_t* gWib   = (uint16_t*)alloc((size_t)NL1 * 384 * H * 2);
    uint16_t* gWhb   = (uint16_t*)alloc((size_t)NL1 * 384 * H * 2);
    uint16_t* Wmt    = (uint16_t*)alloc(H * H * 2);
    uint16_t* mWib   = (uint16_t*)alloc(384 * H * 2);
    uint16_t* mWhb   = (uint16_t*)alloc(384 * H * 2);
    // big region: hj (E*H bf16) later reused for gi/gh (N*384 f32 each)
    size_t need_gigh = 2 * (size_t)N * 384 * 4;
    size_t need_hj   = (size_t)E * H * 2;
    char* big = (char*)alloc(need_hj > need_gigh ? need_hj : need_gigh);
    uint16_t* hjbf = (uint16_t*)big;
    float*    gi   = (float*)big;
    float*    gh   = (float*)(big + (size_t)N * 384 * 4);

    auto nb = [](long n) { return (unsigned)((n + 255) / 256); };
    auto gemm = [&](const uint16_t* A, const int* gidx, int M, const uint16_t* Bt, int Nout,
                    const float* bias, const float* r1c, const float* r1v, int act, int ntOut,
                    float* Out, uint16_t* OutBf, float* Scat, const int* sidx,
                    const float* scale) {
        dim3 g((M + 15) / 16, Nout / 128);
        k_wmma_gemm<<<g, 256, 0, stream>>>(A, gidx, M, Bt, Nout, bias, r1c, r1v, act, ntOut,
                                           Out, OutBf, Scat, sidx, scale);
    };

    // ---- weight prep (bf16, transposed where B is used as x@W) ----
    k_trans128<<<64, 256, 0, stream>>>(W1, W1t);
    k_trans128<<<64, 256, 0, stream>>>(Wg1, Wg1at);        // first 128 rows
    k_trans128<<<64, 256, 0, stream>>>(Wg2, Wg2t);
    k_cvt<<<nb(384 * H), 256, 0, stream>>>(gWi0, gWi0b, 384 * H);   // x@Wi^T -> Bt = Wi
    k_cvt<<<nb(384 * H), 256, 0, stream>>>(gWh0, gWh0b, 384 * H);
    for (int l = 0; l < NL1; ++l)
        k_trans128<<<64, 256, 0, stream>>>(Wc + (size_t)l * H * H, Wct + (size_t)l * H * H);
    k_cvt<<<nb((long)NL1 * 384 * H), 256, 0, stream>>>(gWi, gWib, NL1 * 384 * H);
    k_cvt<<<nb((long)NL1 * 384 * H), 256, 0, stream>>>(gWh, gWhb, NL1 * 384 * H);
    k_trans128<<<64, 256, 0, stream>>>(Wm, Wmt);
    k_cvt<<<nb(384 * H), 256, 0, stream>>>(mWi, mWib, 384 * H);
    k_cvt<<<nb(384 * H), 256, 0, stream>>>(mWh, mWhb, 384 * H);
    k_matvec128<<<1, 128, 0, stream>>>(Wm, aDm, wv);       // wv = Wm @ aDm

    // ---- lin1: x = leaky(x@W1 + b1) ----
    k_cvt<<<nb((long)N * H), 256, 0, stream>>>(x, xbf, N * H);
    gemm(xbf, nullptr, N, W1t, H, b1, nullptr, nullptr, 1, 0,
         xf, xbf, nullptr, nullptr, nullptr);

    // ---- GATEConv ----
    k_rowdot2<<<(N + WPB - 1) / WPB, 256, 0, stream>>>(xf, N, attR, nullptr, dA, nullptr); // dA = x@attR
    // hj = leaky(x[src]@Wg1a + edge_attr*wrow)  -> bf16 [E x H], streamed (NT store)
    gemm(xbf, src, E, Wg1at, H, nullptr, eatt, wrow, 1, 1,
         nullptr, hjbf, nullptr, nullptr, nullptr);
    k_gate_alpha<<<(E + WPB - 1) / WPB, 256, 0, stream>>>(hjbf, E, attL, dA, dst, alphaE);
    k_fill<<<nb(N), 256, 0, stream>>>(segmax, -INFINITY, N);
    k_fill<<<nb(N), 256, 0, stream>>>(segsum, 0.f, N);
    k_seg_max<<<nb(E), 256, 0, stream>>>(alphaE, dst, E, segmax);
    k_seg_expsum<<<nb(E), 256, 0, stream>>>(alphaE, dst, E, segmax, segsum);
    k_seg_norm<<<nb(E), 256, 0, stream>>>(alphaE, dst, E, segsum);
    k_fill<<<nb((long)N * H), 256, 0, stream>>>(hnode, 0.f, N * H);
    // h = segsum((hj@Wg2)*alpha, dst)  — fused scatter in GEMM epilogue
    gemm(hjbf, nullptr, E, Wg2t, H, nullptr, nullptr, nullptr, 0, 0,
         nullptr, nullptr, hnode, dst, alphaE);
    k_elu_bias<<<nb((long)N * H), 256, 0, stream>>>(hnode, bg, N, hbf);
    gemm(hbf, nullptr, N, gWi0b, 384, gbi0, nullptr, nullptr, 0, 0,
         gi, nullptr, nullptr, nullptr, nullptr);
    gemm(xbf, nullptr, N, gWh0b, 384, gbh0, nullptr, nullptr, 0, 0,
         gh, nullptr, nullptr, nullptr, nullptr);
    k_gru<<<nb((long)N * H), 256, 0, stream>>>(gi, gh, xf, xbf, N);

    // ---- GATConv layers ----
    for (int l = 0; l < NL1; ++l) {
        gemm(xbf, nullptr, N, Wct + (size_t)l * H * H, H, nullptr, nullptr, nullptr, 0, 0,
             hsbuf, nullptr, nullptr, nullptr, nullptr);
        k_rowdot2<<<(N + WPB - 1) / WPB, 256, 0, stream>>>(hsbuf, N, aS + l * H, aD + l * H, dA, dB);
        k_edge_alpha<<<nb(E), 256, 0, stream>>>(dA, dB, src, dst, E, alphaE);
        k_fill<<<nb(N), 256, 0, stream>>>(segmax, -INFINITY, N);
        k_fill<<<nb(N), 256, 0, stream>>>(segsum, 0.f, N);
        k_seg_max<<<nb(E), 256, 0, stream>>>(alphaE, dst, E, segmax);
        k_seg_expsum<<<nb(E), 256, 0, stream>>>(alphaE, dst, E, segmax, segsum);
        k_seg_norm<<<nb(E), 256, 0, stream>>>(alphaE, dst, E, segsum);
        k_fill<<<nb((long)N * H), 256, 0, stream>>>(hnode, 0.f, N * H);
        k_edge_scatter<<<(E + WPB - 1) / WPB, 256, 0, stream>>>(hsbuf, alphaE, src, dst, E, hnode);
        k_elu_bias<<<nb((long)N * H), 256, 0, stream>>>(hnode, bc + l * H, N, hbf);
        gemm(hbf, nullptr, N, gWib + (size_t)l * 384 * H, 384, gbi + l * 384,
             nullptr, nullptr, 0, 0, gi, nullptr, nullptr, nullptr, nullptr);
        gemm(xbf, nullptr, N, gWhb + (size_t)l * 384 * H, 384, gbh + l * 384,
             nullptr, nullptr, 0, 0, gh, nullptr, nullptr, nullptr, nullptr);
        k_gru<<<nb((long)N * H), 256, 0, stream>>>(gi, gh, xf, xbf, N);
    }

    // ---- molecule embedding ----
    k_fill<<<nb((long)G * H), 256, 0, stream>>>(outG, 0.f, G * H);
    k_seg_scatter<<<(N + WPB - 1) / WPB, 256, 0, stream>>>(xf, nullptr, batch, N, outG);
    k_relu_bf<<<nb((long)G * H), 256, 0, stream>>>(outG, G * H, outGbf);
    gemm(xbf, nullptr, N, Wmt, H, nullptr, nullptr, nullptr, 0, 0,
         hsbuf, nullptr, nullptr, nullptr, nullptr);                     // hs = x@Wm
    k_rowdot2<<<(N + WPB - 1) / WPB, 256, 0, stream>>>(hsbuf, N, aSm, nullptr, asrc, nullptr);
    for (int t = 0; t < NT; ++t) {
        k_rowdot2<<<(G + WPB - 1) / WPB, 256, 0, stream>>>(outG, G, wv, nullptr, adG, nullptr);
        k_node_alpha<<<nb(N), 256, 0, stream>>>(asrc, adG, batch, N, alphaE);
        k_fill<<<nb(G), 256, 0, stream>>>(segmax, -INFINITY, G);
        k_fill<<<nb(G), 256, 0, stream>>>(segsum, 0.f, G);
        k_seg_max<<<nb(N), 256, 0, stream>>>(alphaE, batch, N, segmax);
        k_seg_expsum<<<nb(N), 256, 0, stream>>>(alphaE, batch, N, segmax, segsum);
        k_seg_norm<<<nb(N), 256, 0, stream>>>(alphaE, batch, N, segsum);
        k_fill<<<nb((long)G * H), 256, 0, stream>>>(hG, 0.f, G * H);
        k_seg_scatter<<<(N + WPB - 1) / WPB, 256, 0, stream>>>(hsbuf, alphaE, batch, N, hG);
        k_elu_bias<<<nb((long)G * H), 256, 0, stream>>>(hG, bm, G, hGbf);
        gemm(hGbf, nullptr, G, mWib, 384, mbi, nullptr, nullptr, 0, 0,
             giG, nullptr, nullptr, nullptr, nullptr);
        gemm(outGbf, nullptr, G, mWhb, 384, mbh, nullptr, nullptr, 0, 0,
             ghG, nullptr, nullptr, nullptr, nullptr);
        k_gru<<<nb((long)G * H), 256, 0, stream>>>(giG, ghG, outG, outGbf, G);
    }
    k_head<<<nb(G), 256, 0, stream>>>(outG, G, W2, b2, Wp1, bp1, Wp2, bp2, (float*)d_out);
}
